// SmoothedMedian_44968307589640
// MI455X (gfx1250) — matrix operand order repaired and checked by reading, not compiled, not run on hardware
//
#include <hip/hip_runtime.h>

// -----------------------------------------------------------------------------
// SmoothedMedian on MI455X (gfx1250, wave32).
// dist(t,s) = sqrt(||x_t||^2 + ||x_s||^2 - 2 x_t.x_s)/C  -> banded Gram matrix.
// Gram tiles via v_wmma_f32_16x16x32_bf16 with split-bf16 (hi+lo):
//   G ~= Ahi*Bhi + Ahi*Blo + Alo*Bhi   (3 WMMAs / 16x16 tile, ~2^-16 rel err).
// Norms computed exactly in f32 from the same (L2-resident) loads, no LDS
// traffic between WMMAs. Branchless distance + masking (EXEC stays all-1s).
// -----------------------------------------------------------------------------

typedef __attribute__((ext_vector_type(16))) __bf16 v16bf;
typedef __attribute__((ext_vector_type(8)))  float  v8f;

#define LDIM 1024
#define CDIM 32
#define TILES_PER_BATCH (LDIM / 16)

__device__ __forceinline__ void split_bf16(float v, __bf16& h, __bf16& l) {
  h = (__bf16)v;                 // v_cvt (RNE)
  l = (__bf16)(v - (float)h);    // residual
}

// A-operand (16-bit A 16x32, own k order, not ISA table order): lane holds row (lane&15);
// elems i = K kb+i, kb = 16*(lane>>4) (contiguous 16 channels, matches packB).
__device__ __forceinline__ void packA(const float f[CDIM], int kb, v16bf& hi, v16bf& lo) {
#pragma unroll
  for (int i = 0; i < 8; ++i) {
    __bf16 h, l;
    split_bf16(f[kb + i], h, l);      hi[i]     = h; lo[i]     = l;
    split_bf16(f[kb + 8 + i], h, l); hi[8 + i] = h; lo[8 + i] = l;
  }
}

// B-operand (16-bit B 32x16, own k order, not ISA table order): lane holds column (lane&15);
// elems i = K koff+i, koff = 16*(lane>>4) (contiguous 16 channels).
__device__ __forceinline__ void packB(const float f[CDIM], int koff, v16bf& hi, v16bf& lo) {
#pragma unroll
  for (int i = 0; i < 16; ++i) {
    __bf16 h, l;
    split_bf16(f[koff + i], h, l);
    hi[i] = h; lo[i] = l;
  }
}

__device__ __forceinline__ void load_row(const float* __restrict__ p,
                                         float f[CDIM], float& norm) {
  float s = 0.f;
#pragma unroll
  for (int i = 0; i < CDIM; ++i) { float v = p[i]; f[i] = v; s += v * v; }
  norm = s;
}

// One wave per 16-row t-tile. 8 waves / block.
__global__ __launch_bounds__(256) void smoothed_tile_kernel(
    const float* __restrict__ X,   // [B, L, C] f32
    const int*   __restrict__ wptr,
    float* __restrict__ pmax,      // [B * L/16] per-tile max of row-maxes
    float* __restrict__ psum) {    // [B * L/16] per-tile sum of row-max^2
  const int w     = *wptr;                       // wave-uniform scalar load
  const int wave  = threadIdx.x >> 5;
  const int tileG = blockIdx.x * (blockDim.x >> 5) + wave;
  const int b     = tileG / TILES_PER_BATCH;
  const int tile  = tileG % TILES_PER_BATCH;
  const int t0    = tile * 16;

  const int lane = threadIdx.x & 31;
  const int nlo  = lane & 15;
  const int hi   = lane >> 4;

  const float* Xb = X + (size_t)b * LDIM * CDIM;

  // ---- A tile: full-row load per lane -> exact f32 norm, split-bf16 operand
  float fa[CDIM]; float a_norm;
  load_row(Xb + (size_t)(t0 + nlo) * CDIM, fa, a_norm);
  v16bf a_hi, a_lo;
  packA(fa, hi * 16, a_hi, a_lo);

  // nt for the 8 rows this lane owns in D: m = r + 8*hi (once per tile)
  float nt8[8];
#pragma unroll
  for (int r = 0; r < 8; ++r) nt8[r] = __shfl(a_norm, r + 8 * hi, 32);

  float cand[8];
#pragma unroll
  for (int r = 0; r < 8; ++r) cand[r] = 0.0f;

  auto body = [&](int j) {
    const int s0 = t0 + j * 16;
    const int s  = s0 + nlo;                     // this lane's column index
    const int rs = min(max(s, 0), LDIM - 1);     // clamped for the load

    float fb[CDIM]; float b_norm;                // full row: local exact norm
    load_row(Xb + (size_t)rs * CDIM, fb, b_norm);
    v16bf b_hi, b_lo;
    packB(fb, hi * 16, b_hi, b_lo);

    v8f acc = {0.f, 0.f, 0.f, 0.f, 0.f, 0.f, 0.f, 0.f};
    acc = __builtin_amdgcn_wmma_f32_16x16x32_bf16(false, a_hi, false, b_hi,
                                                  (short)0, acc, false, false);
    acc = __builtin_amdgcn_wmma_f32_16x16x32_bf16(false, a_hi, false, b_lo,
                                                  (short)0, acc, false, false);
    acc = __builtin_amdgcn_wmma_f32_16x16x32_bf16(false, a_lo, false, b_hi,
                                                  (short)0, acc, false, false);

    // D layout: lane holds column n = nlo; VGPR r holds row m = r + 8*hi.
#pragma unroll
    for (int r = 0; r < 8; ++r) {
      const int t = t0 + r + 8 * hi;
      const bool valid = (s >= 0) && (s < LDIM) && (s >= t - w) && (s <= t + w);
      float sq = nt8[r] + b_norm - 2.0f * acc[r];
      float d  = __builtin_amdgcn_sqrtf(fmaxf(sq, 0.0f)) * (1.0f / (float)CDIM);
      cand[r]  = fmaxf(cand[r], valid ? d : 0.0f);   // branchless mask
    }
  };

  if (w == 32) {
    // Harness case: fully unrolled -> 15 static WMMAs, software-pipelined.
#pragma unroll
    for (int j = -2; j <= 2; ++j) body(j);
  } else {
    const int jlo = -((w + 15) >> 4);
    const int jhi =  (15 + w) >> 4;
    for (int j = jlo; j <= jhi; ++j) body(j);
  }

  // ---- reductions (deterministic shuffle trees) ----
  float pm = 0.0f, ps = 0.0f;
#pragma unroll
  for (int r = 0; r < 8; ++r) {
    float v = cand[r];
    v = fmaxf(v, __shfl_xor(v, 1, 32));
    v = fmaxf(v, __shfl_xor(v, 2, 32));
    v = fmaxf(v, __shfl_xor(v, 4, 32));
    v = fmaxf(v, __shfl_xor(v, 8, 32));  // row max over the 16 columns
    pm = fmaxf(pm, v);
    ps += v * v;
  }
  pm = fmaxf(pm, __shfl_xor(pm, 16, 32));  // combine rows 0..7 with 8..15
  ps = ps + __shfl_xor(ps, 16, 32);

  if (lane == 0) { pmax[tileG] = pm; psum[tileG] = ps; }
}

// One wave per batch: fold 64 tile partials deterministically.
__global__ __launch_bounds__(32) void smoothed_reduce_kernel(
    const float* __restrict__ pmax, const float* __restrict__ psum,
    float* __restrict__ out, int B) {
  const int b    = blockIdx.x;
  const int lane = threadIdx.x;
  float m = 0.0f, s = 0.0f;
  for (int i = lane; i < TILES_PER_BATCH; i += 32) {
    m = fmaxf(m, pmax[b * TILES_PER_BATCH + i]);
    s += psum[b * TILES_PER_BATCH + i];
  }
#pragma unroll
  for (int off = 1; off < 32; off <<= 1) {
    m = fmaxf(m, __shfl_xor(m, off, 32));
    s = s + __shfl_xor(s, off, 32);
  }
  if (lane == 0) { out[b] = m; out[B + b] = s; }
}

extern "C" void kernel_launch(void* const* d_in, const int* in_sizes, int n_in,
                              void* d_out, int out_size, void* d_ws, size_t ws_size,
                              hipStream_t stream) {
  (void)in_sizes; (void)n_in; (void)out_size; (void)ws_size;
  const float* X    = (const float*)d_in[0];
  const int*   wptr = (const int*)d_in[1];
  float*       out  = (float*)d_out;

  const int B = 32;
  const int tiles = B * TILES_PER_BATCH;        // 2048 waves
  float* pmax = (float*)d_ws;                    // 2048 floats
  float* psum = pmax + tiles;                    // 2048 floats

  smoothed_tile_kernel<<<dim3(tiles / 8), dim3(256), 0, stream>>>(X, wptr, pmax, psum);
  smoothed_reduce_kernel<<<dim3(B), dim3(32), 0, stream>>>(pmax, psum, out, B);
}
